// HybridVPRLoss_13683765805398
// MI455X (gfx1250) — matrix operand order repaired and checked by reading, not compiled, not run on hardware
//
#include <hip/hip_runtime.h>
#include <hip/hip_bf16.h>

#define NROWS 4096
#define DDIM  512
#define TTRIP 4096

// contrastive tiling
#define ROWS_PER_BLOCK 128            // 8 waves x 16 rows
#define COLS_PER_BLOCK 512            // 32 column tiles of 16
#define COL_TILES      (COLS_PER_BLOCK / 16)
#define LDS_BSTRIDE    520            // padded row stride (halfs) for bank spread

typedef _Float16 v8h  __attribute__((ext_vector_type(8)));
typedef _Float16 v16h __attribute__((ext_vector_type(16)));
typedef float    v8f  __attribute__((ext_vector_type(8)));

// ---------------- zero scalar + per-row accumulators ----------------
// layout in gScratch (floats): [0..3] gAccum, [4..4+4096) pos, then tot, then cnt
#define SCRATCH_FLOATS (4 + 3 * NROWS)
__global__ void zero_scratch_kernel(float* g) {
    int i = blockIdx.x * blockDim.x + threadIdx.x;
    if (i < SCRATCH_FLOATS) g[i] = 0.0f;
}

// ---------------- normalize rows -> f16 ----------------
__global__ void normalize_kernel(const float* __restrict__ emb,
                                 _Float16* __restrict__ nemb) {
    int row = blockIdx.x;
    const float* r = emb + (size_t)row * DDIM;
    float ss = 0.0f;
    for (int c = threadIdx.x; c < DDIM; c += blockDim.x) {
        float v = r[c];
        ss += v * v;
    }
    for (int off = 16; off >= 1; off >>= 1) ss += __shfl_xor(ss, off, 32);
    __shared__ float red[8];
    int wid = threadIdx.x >> 5, lane = threadIdx.x & 31;
    if (lane == 0) red[wid] = ss;
    __syncthreads();
    if (threadIdx.x == 0) {
        float t = 0.0f;
        int nw = blockDim.x >> 5;
        for (int w = 0; w < nw; ++w) t += red[w];
        red[0] = t;
    }
    __syncthreads();
    float norm  = sqrtf(red[0]);
    float scale = 1.0f / fmaxf(norm, 1e-8f);
    for (int c = threadIdx.x; c < DDIM; c += blockDim.x)
        nemb[(size_t)row * DDIM + c] = (_Float16)(r[c] * scale);
}

// ---------------- fused WMMA contrastive partial sums ----------------
// grid: (8 col-groups, 32 row-groups), block: 256 threads (8 wave32s).
// Block covers rows [rowGroup*128, +128) x cols [colGroup*512, +512).
// Each wave owns one 16-row strip (A cached in regs for all K);
// the 16x512 f16 B panel for each column tile is staged once in LDS and
// shared by all 8 waves -> 8x less L2 traffic than per-wave global streaming.
__global__ void __launch_bounds__(256)
contrastive_kernel(const _Float16* __restrict__ nemb,
                   const int* __restrict__ labels,
                   float* __restrict__ gPos,
                   float* __restrict__ gTot,
                   float* __restrict__ gCnt) {
    __shared__ _Float16 sB[16 * LDS_BSTRIDE];   // 16.25 KB B panel
    __shared__ int sLab[COLS_PER_BLOCK];        // 2 KB column labels

    const int tid      = threadIdx.x;
    const int colGroup = blockIdx.x;
    const int rowGroup = blockIdx.y;
    const int colBase  = colGroup * COLS_PER_BLOCK;
    const int wave     = tid >> 5;
    const int lane     = tid & 31;
    const int m        = lane & 15;   // A row / B col within tile
    const int halfSel  = lane >> 4;
    const int i0       = rowGroup * ROWS_PER_BLOCK + wave * 16;

    // stage this block's column labels
    sLab[tid]       = labels[colBase + tid];
    sLab[tid + 256] = labels[colBase + tid + 256];

    // ---- preload A fragments (rows i0..i0+15, all K=512) into registers ----
    const _Float16* arow = nemb + (size_t)(i0 + m) * DDIM;
    v16h afrag[16];
    #pragma unroll
    for (int ks = 0; ks < 16; ++ks) {
        const int kb = ks * 32 + halfSel * 8;
        v8h c0 = *(const v8h*)(arow + kb);        // K = kb .. kb+7
        v8h c1 = *(const v8h*)(arow + kb + 16);   // K = kb+16 .. kb+23
        afrag[ks] = __builtin_shufflevector(c0, c1,
            0,1,2,3,4,5,6,7, 8,9,10,11,12,13,14,15);
    }

    int rowLab[8];
    #pragma unroll
    for (int r = 0; r < 8; ++r) rowLab[r] = labels[i0 + r + 8 * halfSel];

    float pPos[8] = {}, pTot[8] = {}, pCnt[8] = {};
    const float invT = 1.0f / 0.07f;

    // cooperative staging: 16 rows x 512 halfs = 1024 chunks of 16B;
    // 256 threads x 4 chunks. chunk c: row = c/64, 16B-piece = c%64.
    uint4 st[4];
    {   // prefetch tile 0
        const int tcb = colBase;  // tile 0 columns
        #pragma unroll
        for (int q = 0; q < 4; ++q) {
            int c = tid + q * 256, row = c >> 6, w = c & 63;
            st[q] = *(const uint4*)(nemb + (size_t)(tcb + row) * DDIM + w * 8);
        }
    }

    for (int jt = 0; jt < COL_TILES; ++jt) {
        __syncthreads();                       // previous tile fully consumed
        #pragma unroll
        for (int q = 0; q < 4; ++q) {          // regs -> LDS
            int c = tid + q * 256, row = c >> 6, w = c & 63;
            *(uint4*)&sB[row * LDS_BSTRIDE + w * 8] = st[q];
        }
        __syncthreads();                       // panel visible
        if (jt + 1 < COL_TILES) {              // prefetch next tile
            const int tcb = colBase + (jt + 1) * 16;
            #pragma unroll
            for (int q = 0; q < 4; ++q) {
                int c = tid + q * 256, row = c >> 6, w = c & 63;
                st[q] = *(const uint4*)(nemb + (size_t)(tcb + row) * DDIM + w * 8);
            }
        }

        // ---- 16x16 tile: 16 WMMAs fed from LDS ----
        const _Float16* brow = &sB[m * LDS_BSTRIDE];
        v8f acc = {};
        #pragma unroll
        for (int ks = 0; ks < 16; ++ks) {
            const int kb = ks * 32 + halfSel * 8;
            v8h b0 = *(const v8h*)(brow + kb);       // K = kb .. kb+7
            v8h b1 = *(const v8h*)(brow + kb + 16);  // K = kb+16 .. kb+23
            v16h bfrag = __builtin_shufflevector(b0, b1,
                0,1,2,3,4,5,6,7, 8,9,10,11,12,13,14,15);
            acc = __builtin_amdgcn_wmma_f32_16x16x32_f16(
                false, afrag[ks], false, bfrag, (short)0, acc, false, false);
        }

        const int col    = colBase + jt * 16 + m;
        const int colLab = sLab[jt * 16 + m];
        #pragma unroll
        for (int r = 0; r < 8; ++r) {
            const int rowIdx = i0 + r + 8 * halfSel;
            float e = expf(acc[r] * invT);
            bool msk = (rowLab[r] == colLab) && (rowIdx != col);
            pTot[r] += e;
            pPos[r] += msk ? e : 0.0f;
            pCnt[r] += msk ? 1.0f : 0.0f;
        }
    }

    // reduce over the 16 lanes of each half; lanes 0/16 publish 8 rows each
    #pragma unroll
    for (int r = 0; r < 8; ++r) {
        float a = pPos[r], b = pTot[r], c = pCnt[r];
        for (int off = 8; off >= 1; off >>= 1) {
            a += __shfl_xor(a, off, 16);
            b += __shfl_xor(b, off, 16);
            c += __shfl_xor(c, off, 16);
        }
        if (m == 0) {
            int rowIdx = i0 + r + 8 * halfSel;
            atomicAdd(&gPos[rowIdx], a);
            atomicAdd(&gTot[rowIdx], b);
            atomicAdd(&gCnt[rowIdx], c);
        }
    }
}

// ---------------- per-row loss + reduction ----------------
__global__ void rowloss_kernel(const float* __restrict__ gPos,
                               const float* __restrict__ gTot,
                               const float* __restrict__ gCnt,
                               float* __restrict__ gAccum) {
    int i = blockIdx.x * blockDim.x + threadIdx.x;   // 4096 threads
    float pos = gPos[i], tot = gTot[i], cnt = gCnt[i];
    float loss  = -logf(pos / (tot + 1e-8f) + 1e-8f);
    bool  valid = cnt > 0.0f;
    float l = valid ? loss : 0.0f;
    float v = valid ? 1.0f : 0.0f;
    for (int off = 16; off >= 1; off >>= 1) {
        l += __shfl_xor(l, off, 32);
        v += __shfl_xor(v, off, 32);
    }
    if ((threadIdx.x & 31) == 0) {
        atomicAdd(&gAccum[0], l);
        atomicAdd(&gAccum[1], v);
    }
}

// ---------------- triplet loss: one wave per triplet ----------------
__global__ void triplet_kernel(const float* __restrict__ emb,
                               const int* __restrict__ trip,
                               float* __restrict__ gAccum) {
    const int wave = threadIdx.x >> 5;
    const int lane = threadIdx.x & 31;
    const int t = blockIdx.x * 8 + wave;
    const int ia = trip[t * 3 + 0];
    const int ip = trip[t * 3 + 1];
    const int in_ = trip[t * 3 + 2];
    const float* a = emb + (size_t)ia * DDIM;
    const float* p = emb + (size_t)ip * DDIM;
    const float* g = emb + (size_t)in_ * DDIM;
    float sp = 0.0f, sn = 0.0f;
    for (int c = lane; c < DDIM; c += 32) {
        float av = a[c];
        float dp = av - p[c] + 1e-6f;
        float dn = av - g[c] + 1e-6f;
        sp += dp * dp;
        sn += dn * dn;
    }
    for (int off = 16; off >= 1; off >>= 1) {
        sp += __shfl_xor(sp, off, 32);
        sn += __shfl_xor(sn, off, 32);
    }
    if (lane == 0) {
        float h = fmaxf(sqrtf(sp) - sqrtf(sn) + 0.5f, 0.0f);
        atomicAdd(&gAccum[2], h);
    }
}

// ---------------- finalize: 3 scalars ----------------
__global__ void finalize_kernel(const float* __restrict__ gAccum,
                                float* __restrict__ out) {
    if (threadIdx.x == 0) {
        float num = gAccum[0], cnt = gAccum[1], ts = gAccum[2];
        float cont = (cnt > 0.0f) ? num / fmaxf(cnt, 1.0f) : 0.0f;
        float trip = ts / (float)TTRIP;
        out[0] = 1.0f * cont + 0.3f * trip;
        out[1] = cont;
        out[2] = trip;
    }
}

extern "C" void kernel_launch(void* const* d_in, const int* in_sizes, int n_in,
                              void* d_out, int out_size, void* d_ws, size_t ws_size,
                              hipStream_t stream) {
    const float* emb    = (const float*)d_in[0];   // (4096, 512) f32
    const int*   labels = (const int*)d_in[1];     // (4096,)     i32
    const int*   trips  = (const int*)d_in[2];     // (4096, 3)   i32
    float* out = (float*)d_out;                    // (total, cont, trip)

    // workspace: [0, 4MB) f16 normalized embeddings; then float scratch
    _Float16* nemb     = (_Float16*)d_ws;
    float*    gScratch = (float*)((char*)d_ws + (size_t)NROWS * DDIM * sizeof(_Float16));
    float*    gAccum   = gScratch;                 // [0..3]
    float*    gPos     = gScratch + 4;             // [4096]
    float*    gTot     = gPos + NROWS;             // [4096]
    float*    gCnt     = gTot + NROWS;             // [4096]

    zero_scratch_kernel<<<(SCRATCH_FLOATS + 255) / 256, 256, 0, stream>>>(gScratch);
    normalize_kernel<<<NROWS, 128, 0, stream>>>(emb, nemb);
    {
        dim3 grid(NROWS * 16 / (ROWS_PER_BLOCK * COLS_PER_BLOCK) * 16, 1, 1); // placeholder avoided below
    }
    {
        dim3 grid(NROWS / COLS_PER_BLOCK * 1, 1, 1); // not used
    }
    dim3 cgrid(NROWS / COLS_PER_BLOCK, NROWS / ROWS_PER_BLOCK, 1); // (8, 32)
    contrastive_kernel<<<cgrid, 256, 0, stream>>>(nemb, labels, gPos, gTot, gCnt);
    rowloss_kernel<<<NROWS / 256, 256, 0, stream>>>(gPos, gTot, gCnt, gAccum);
    triplet_kernel<<<TTRIP / 8, 256, 0, stream>>>(emb, trips, gAccum);
    finalize_kernel<<<1, 32, 0, stream>>>(gAccum, out);
}
